// TC_47313359732764
// MI455X (gfx1250) — matrix operand order repaired and checked by reading, not compiled, not run on hardware
//
#include <hip/hip_runtime.h>
#include <hip/hip_bf16.h>
#include <math.h>
#include <cstdint>

typedef __attribute__((ext_vector_type(16))) __bf16 v16bf;
typedef __attribute__((ext_vector_type(8)))  __bf16 v8bf;
typedef __attribute__((ext_vector_type(8)))  float  v8f;

#define T_STEPS 49
#define BATCH   1024
#define CDIM    128
#define NCLS    10
#define HDIM    64
#define LDSROW  136   // 128 + 8 bf16 pad (16B/row) to break LDS bank conflicts

union V16 { v16bf v; v8bf h[2]; };

#if defined(__gfx1250__) && __has_builtin(__builtin_amdgcn_tensor_load_to_lds) && __has_builtin(__builtin_amdgcn_s_wait_tensorcnt)
#define USE_TDM 1
#endif

#ifdef USE_TDM
typedef __attribute__((ext_vector_type(4))) unsigned int v4u;
typedef __attribute__((ext_vector_type(4))) int v4i;
typedef __attribute__((ext_vector_type(8))) int v8i;
union G0d { unsigned long long q[2]; v4u v; };
union G1d { unsigned long long q[4]; v8i v; };

// TDM: DMA a 16-row x 256B tile (bf16, contiguous rows) into LDS with a 16B pad
// inserted after every 256B row (reproducing the LDSROW=136 layout).
// D# layout per CDNA5 ISA ch.8: group0 = {count=1 | lds_addr<<32 , global_addr | type2};
// group1 = {data_size=8B, pad_enable, pad_interval=64dw, pad_amount=4dw, tensor_dim0=32,
//           tensor_dim1=16, tile_dim0=32, tile_dim1=16, dim0_stride=32}.
__device__ __forceinline__ void tdm_load_tile(const __bf16* gsrc, unsigned lds_byte_off) {
  G0d g0; G1d g1;
  const unsigned long long ga = (unsigned long long)(uintptr_t)gsrc;
  g0.q[0] = 1ull | ((unsigned long long)lds_byte_off << 32);
  g0.q[1] = (ga & 0x01FFFFFFFFFFFFFFull) | (2ull << 62);         // type=2 ("image")
  g1.q[0] = (3ull << 16)      // data_size = 8 bytes
          | (1ull << 20)      // pad_enable
          | (5ull << 22)      // pad_interval: 2^(5+1) = 64 DWORDs (= 256B row)
          | (3ull << 25)      // pad_amount: 4 DWORDs (= 16B)
          | (32ull << 48);    // tensor_dim0 = 32 units (low 16 bits)
  g1.q[1] = (16ull << 16)     // tensor_dim1 = 16 (bits 111:80)
          | (32ull << 48);    // tile_dim0  = 32 (bits 127:112)
  g1.q[2] = 16ull             // tile_dim1  = 16 (bits 143:128)
          | (32ull << 32);    // tensor_dim0_stride = 32 (bits 191:160)
  g1.q[3] = 0;
  const v4i z4 = (v4i)0;
#if __clang_major__ >= 23
  const v8i z8 = (v8i)0;
  __builtin_amdgcn_tensor_load_to_lds(g0.v, g1.v, z4, z4, z8, 0);
#else
  __builtin_amdgcn_tensor_load_to_lds(g0.v, g1.v, z4, z4, 0);
#endif
}
#endif // USE_TDM

__device__ __forceinline__ void split1(float x, __bf16* hi, __bf16* lo) {
  __bf16 h = (__bf16)x;
  *hi = h;
  *lo = (__bf16)(x - (float)h);
}

// bf16x3 error-compensated f32 GEMM step: (Ahi+Alo)(Bhi+Blo) ~= AhiBhi + AhiBlo + AloBhi
__device__ __forceinline__ v8f wmma3(v8f acc, const V16& ahi, const V16& alo,
                                     const V16& bhi, const V16& blo) {
  acc = __builtin_amdgcn_wmma_f32_16x16x32_bf16(false, ahi.v, false, bhi.v, (short)0, acc, false, false);
  acc = __builtin_amdgcn_wmma_f32_16x16x32_bf16(false, ahi.v, false, blo.v, (short)0, acc, false, false);
  acc = __builtin_amdgcn_wmma_f32_16x16x32_bf16(false, alo.v, false, bhi.v, (short)0, acc, false, false);
  return acc;
}

__device__ __forceinline__ float red_max16(float x) {
#pragma unroll
  for (int m = 1; m < 16; m <<= 1) x = fmaxf(x, __shfl_xor(x, m, 32));
  return x;
}
__device__ __forceinline__ float red_add16(float x) {
#pragma unroll
  for (int m = 1; m < 16; m <<= 1) x += __shfl_xor(x, m, 32);
  return x;
}

// A fragment (16x32 bf16): lane holds row M=lane%16; elems 0..7 -> K=kb+8*half+e,
// elems 8..15 -> K=kb+16+8*half+(e-8). Source rows are contiguous bf16.
__device__ __forceinline__ void load_afrag(const __bf16* row, int kb, int half, V16* f) {
  f->h[0] = *reinterpret_cast<const v8bf*>(row + kb + 8 * half);
  f->h[1] = *reinterpret_cast<const v8bf*>(row + kb + 16 + 8 * half);
}

// B fragment (32x16 bf16) from LDS tile stored [N=16 rows][128 K] (padded rows):
// lane holds col N=lane%16; elems 0..7 -> K=kb+8*half+e, elems 8..15 -> K=kb+16+8*half+(e-8).
__device__ __forceinline__ void load_bfrag(const __bf16* lds, int n, int kb, int half, V16* f) {
  const __bf16* p = lds + n * LDSROW + kb + 8 * half;
  f->h[0] = *reinterpret_cast<const v8bf*>(p);
  f->h[1] = *reinterpret_cast<const v8bf*>(p + 16);
}

// ---------------- Phase A: c_t = mean_t(features_aug1), plus bf16 hi/lo split ----------------
__global__ __launch_bounds__(256) void k_ct(const float* __restrict__ f1, float* __restrict__ ct,
                                            __bf16* __restrict__ ct_hi, __bf16* __restrict__ ct_lo) {
  int idx = blockIdx.x * 256 + threadIdx.x;        // B*C = 131072
  int b = idx >> 7, c = idx & 127;
  const float* p = f1 + (size_t)b * T_STEPS * CDIM + c;
  float s = 0.f;
#pragma unroll
  for (int t = 0; t < T_STEPS; ++t) s += p[t * CDIM];
  s *= (1.0f / 49.0f);
  ct[idx] = s;
  split1(s, &ct_hi[idx], &ct_lo[idx]);
}

__global__ void k_zero(float* p) { if (threadIdx.x == 0 && blockIdx.x == 0) *p = 0.f; }

// ---------------- Phase B: pred[t] = c_t @ Wk_w[t]^T + Wk_b[t], stored as split bf16 ----------------
// grid = (8 rowblocks, 49 t), block = 256 (8 waves x 16 rows each)
__global__ __launch_bounds__(256) void k_pred(const __bf16* __restrict__ ct_hi, const __bf16* __restrict__ ct_lo,
                                              const float* __restrict__ Wk_w, const float* __restrict__ Wk_b,
                                              __bf16* __restrict__ pred_hi, __bf16* __restrict__ pred_lo) {
  __shared__ alignas(16) __bf16 ldsH[16 * LDSROW];
  __shared__ alignas(16) __bf16 ldsL[16 * LDSROW];
  const int t = blockIdx.y;
  const int wave = threadIdx.x >> 5;
  const int lane = threadIdx.x & 31;
  const int half = lane >> 4, m = lane & 15;
  const int row0 = blockIdx.x * 128 + wave * 16;

  V16 aH[4], aL[4];
  const __bf16* arH = ct_hi + (size_t)(row0 + m) * CDIM;
  const __bf16* arL = ct_lo + (size_t)(row0 + m) * CDIM;
#pragma unroll
  for (int k = 0; k < 4; ++k) { load_afrag(arH, 32 * k, half, &aH[k]); load_afrag(arL, 32 * k, half, &aL[k]); }

  const int jr = threadIdx.x >> 4;        // 0..15 staging row
  const int c0 = (threadIdx.x & 15) * 8;  // 0..120 staging col base
  for (int cb = 0; cb < 8; ++cb) {
    const int col0 = cb * 16;
    const float* wsrc = Wk_w + ((size_t)t * CDIM + col0 + jr) * CDIM + c0;
#pragma unroll
    for (int e = 0; e < 8; ++e)
      split1(wsrc[e], &ldsH[jr * LDSROW + c0 + e], &ldsL[jr * LDSROW + c0 + e]);
    __syncthreads();

    v8f acc = {0.f, 0.f, 0.f, 0.f, 0.f, 0.f, 0.f, 0.f};
#pragma unroll
    for (int k = 0; k < 4; ++k) {
      V16 bH, bL;
      load_bfrag(ldsH, m, 32 * k, half, &bH);
      load_bfrag(ldsL, m, 32 * k, half, &bL);
      acc = wmma3(acc, aH[k], aL[k], bH, bL);
    }
    const int gcol = col0 + m;
    const float bias = Wk_b[t * CDIM + gcol];
#pragma unroll
    for (int r = 0; r < 8; ++r) {
      const int rr = row0 + r + 8 * half;
      const float v = acc[r] + bias;
      const size_t off = ((size_t)t * BATCH + rr) * CDIM + gcol;
      split1(v, &pred_hi[off], &pred_lo[off]);
    }
    __syncthreads();
  }
}

// ---------------- Phase C: total[t]=E_t P_t^T fused with online log-softmax diag (NCE) ----------------
// grid = (8 rowblocks, 49 t), block = 256. Each wave owns 16 complete rows and
// streams all 64 column tiles; never materializes the (49,1024,1024) tensor.
// B tiles are staged into LDS by the Tensor Data Mover (pad fields reproduce the
// padded LDSROW layout); waves wait on TENSORcnt + workgroup barrier.
__global__ __launch_bounds__(256) void k_nce(const float* __restrict__ f2,
                                             const __bf16* __restrict__ pred_hi, const __bf16* __restrict__ pred_lo,
                                             float* __restrict__ nce_acc) {
  __shared__ alignas(16) __bf16 ldsH[16 * LDSROW];
  __shared__ alignas(16) __bf16 ldsL[16 * LDSROW];
  const int t = blockIdx.y;
  const int wave = threadIdx.x >> 5;
  const int lane = threadIdx.x & 31;
  const int half = lane >> 4, m = lane & 15;
  const int row0 = blockIdx.x * 128 + wave * 16;

  // A = encode[t] rows (f32 in features_aug2 layout (B,T,C)); split to bf16 hi/lo on the fly.
  V16 aH[4], aL[4];
  {
    const float* arow = f2 + ((size_t)(row0 + m) * T_STEPS + t) * CDIM;
#pragma unroll
    for (int k = 0; k < 4; ++k) {
      const int b0 = 32 * k + 8 * half;
      const int b1 = 32 * k + 16 + 8 * half;
#pragma unroll
      for (int e = 0; e < 8; ++e) {
        __bf16 hi, lo;
        split1(arow[b0 + e], &hi, &lo); aH[k].v[e] = hi;     aL[k].v[e] = lo;
        split1(arow[b1 + e], &hi, &lo); aH[k].v[8 + e] = hi; aL[k].v[8 + e] = lo;
      }
    }
  }

  float mrun[8], srun[8], dsum[8];
#pragma unroll
  for (int r = 0; r < 8; ++r) { mrun[r] = -INFINITY; srun[r] = 0.f; dsum[r] = 0.f; }

#ifdef USE_TDM
  const unsigned ldsH_off = (unsigned)(uintptr_t)&ldsH[0];
  const unsigned ldsL_off = (unsigned)(uintptr_t)&ldsL[0];
#else
  const int jr = threadIdx.x >> 4;
  const int c0 = (threadIdx.x & 15) * 8;
#endif

  for (int cb = 0; cb < 64; ++cb) {
    const int col0 = cb * 16;
#ifdef USE_TDM
    if (wave == 0) {
      tdm_load_tile(pred_hi + ((size_t)t * BATCH + col0) * CDIM, ldsH_off);
      tdm_load_tile(pred_lo + ((size_t)t * BATCH + col0) * CDIM, ldsL_off);
      __builtin_amdgcn_s_wait_tensorcnt(0);
    }
    __syncthreads();
#else
    // fallback: manual staging of 16 pred rows (already bf16 split) into padded LDS
    const size_t srcoff = ((size_t)t * BATCH + col0 + jr) * CDIM + c0;
    *reinterpret_cast<v8bf*>(&ldsH[jr * LDSROW + c0]) = *reinterpret_cast<const v8bf*>(pred_hi + srcoff);
    *reinterpret_cast<v8bf*>(&ldsL[jr * LDSROW + c0]) = *reinterpret_cast<const v8bf*>(pred_lo + srcoff);
    if (cb < 63) {
      __builtin_prefetch(pred_hi + srcoff + 16 * CDIM, 0, 1);
      __builtin_prefetch(pred_lo + srcoff + 16 * CDIM, 0, 1);
    }
    __syncthreads();
#endif

    v8f acc = {0.f, 0.f, 0.f, 0.f, 0.f, 0.f, 0.f, 0.f};
#pragma unroll
    for (int k = 0; k < 4; ++k) {
      V16 bH, bL;
      load_bfrag(ldsH, m, 32 * k, half, &bH);
      load_bfrag(ldsL, m, 32 * k, half, &bL);
      acc = wmma3(acc, aH[k], aL[k], bH, bL);
    }

    // online logsumexp per row (rows live in one VGPR across a 16-lane half)
#pragma unroll
    for (int r = 0; r < 8; ++r) {
      const float x = acc[r];
      const int rr = row0 + r + 8 * half;
      const int cc = col0 + m;
      if (rr == cc) dsum[r] += x;
      const float tm = red_max16(x);
      const float ts = red_add16(__expf(x - tm));
      const float nm = fmaxf(mrun[r], tm);
      srun[r] = srun[r] * __expf(mrun[r] - nm) + ts * __expf(tm - nm);
      mrun[r] = nm;
    }
    __syncthreads();
  }

  float contrib = 0.f;
#pragma unroll
  for (int r = 0; r < 8; ++r)
    contrib += red_add16(dsum[r]) - (mrun[r] + __logf(srun[r]));
  if (m == 0) atomicAdd(nce_acc, contrib);   // one add per 16-lane half (8 rows each)
}

// ---------------- Phase D: projection head h, logits (tiny, VALU) ----------------
__global__ __launch_bounds__(256) void k_head(const float* __restrict__ ct,
                                              const float* __restrict__ proj_w, const float* __restrict__ proj_b,
                                              const float* __restrict__ logits_w, const float* __restrict__ logits_b,
                                              float* __restrict__ h, float* __restrict__ out_logits) {
  const int idx = blockIdx.x * 256 + threadIdx.x;     // 65536 + 10240 = 75776
  if (idx < BATCH * HDIM) {
    const int b = idx / HDIM, j = idx % HDIM;
    const float* a = ct + (size_t)b * CDIM;
    const float* w = proj_w + (size_t)j * CDIM;
    float s = proj_b[j];
#pragma unroll 8
    for (int c = 0; c < CDIM; ++c) s = fmaf(a[c], w[c], s);
    h[idx] = s;
  } else {
    const int i2 = idx - BATCH * HDIM;
    const int b = i2 / NCLS, n = i2 % NCLS;
    const float* a = ct + (size_t)b * CDIM;
    const float* w = logits_w + (size_t)n * CDIM;
    float s = logits_b[n];
#pragma unroll 8
    for (int c = 0; c < CDIM; ++c) s = fmaf(a[c], w[c], s);
    out_logits[i2] = s;
  }
}

__global__ __launch_bounds__(256) void k_bnstats(const float* __restrict__ h,
                                                 float* __restrict__ mu, float* __restrict__ var) {
  const int j = blockIdx.x;                     // 64 columns
  __shared__ float ssum[256], ssq[256];
  float s = 0.f, q = 0.f;
  for (int b = threadIdx.x; b < BATCH; b += 256) {
    const float v = h[(size_t)b * HDIM + j];
    s += v; q += v * v;
  }
  ssum[threadIdx.x] = s; ssq[threadIdx.x] = q;
  __syncthreads();
  for (int st = 128; st > 0; st >>= 1) {
    if (threadIdx.x < st) { ssum[threadIdx.x] += ssum[threadIdx.x + st]; ssq[threadIdx.x] += ssq[threadIdx.x + st]; }
    __syncthreads();
  }
  if (threadIdx.x == 0) {
    const float mmu = ssum[0] * (1.0f / BATCH);
    mu[j] = mmu;
    var[j] = ssq[0] * (1.0f / BATCH) - mmu * mmu;   // biased (BN training mode)
  }
}

__global__ __launch_bounds__(256) void k_bnapply(const float* __restrict__ h,
                                                 const float* __restrict__ mu, const float* __restrict__ var,
                                                 const float* __restrict__ gamma, const float* __restrict__ beta,
                                                 float* __restrict__ out) {
  const int idx = blockIdx.x * 256 + threadIdx.x;   // 65536
  const int j = idx & (HDIM - 1);
  const float v = (h[idx] - mu[j]) * rsqrtf(var[j] + 1e-5f) * gamma[j] + beta[j];
  out[idx] = fmaxf(v, 0.f);
}

__global__ void k_fin(const float* acc, float* out0) {
  if (threadIdx.x == 0 && blockIdx.x == 0)
    out0[0] = (*acc) * (-1.0f / (float)(BATCH * T_STEPS));
}

extern "C" void kernel_launch(void* const* d_in, const int* in_sizes, int n_in,
                              void* d_out, int out_size, void* d_ws, size_t ws_size,
                              hipStream_t stream) {
  const float* f1       = (const float*)d_in[0];   // (B,T,C)
  const float* f2       = (const float*)d_in[1];   // (B,T,C)
  const float* Wk_w     = (const float*)d_in[2];   // (T,C,C)
  const float* Wk_b     = (const float*)d_in[3];   // (T,C)
  const float* proj_w   = (const float*)d_in[4];   // (64,128)
  const float* proj_b   = (const float*)d_in[5];
  const float* bn_gamma = (const float*)d_in[6];
  const float* bn_beta  = (const float*)d_in[7];
  const float* logits_w = (const float*)d_in[8];   // (10,128)
  const float* logits_b = (const float*)d_in[9];
  float* out = (float*)d_out;                      // [nce(1) | proj(1024*64) | logits(1024*10)]

  char* w = (char*)d_ws;
  size_t off = 0;
  auto alloc = [&](size_t bytes) -> void* {
    void* p = w + off; off += (bytes + 255) & ~(size_t)255; return p;
  };
  float*  ct      = (float*) alloc((size_t)BATCH * CDIM * 4);
  __bf16* ct_hi   = (__bf16*)alloc((size_t)BATCH * CDIM * 2);
  __bf16* ct_lo   = (__bf16*)alloc((size_t)BATCH * CDIM * 2);
  __bf16* pred_hi = (__bf16*)alloc((size_t)T_STEPS * BATCH * CDIM * 2);
  __bf16* pred_lo = (__bf16*)alloc((size_t)T_STEPS * BATCH * CDIM * 2);
  float*  hbuf    = (float*) alloc((size_t)BATCH * HDIM * 4);
  float*  mu      = (float*) alloc(HDIM * 4);
  float*  var     = (float*) alloc(HDIM * 4);
  float*  nacc    = (float*) alloc(256);
  (void)ws_size; (void)in_sizes; (void)n_in; (void)out_size;

  k_ct  <<<(BATCH * CDIM) / 256, 256, 0, stream>>>(f1, ct, ct_hi, ct_lo);
  k_zero<<<1, 64, 0, stream>>>(nacc);
  k_pred<<<dim3(8, T_STEPS), 256, 0, stream>>>(ct_hi, ct_lo, Wk_w, Wk_b, pred_hi, pred_lo);
  k_nce <<<dim3(8, T_STEPS), 256, 0, stream>>>(f2, pred_hi, pred_lo, nacc);
  k_head<<<(BATCH * (HDIM + NCLS)) / 256, 256, 0, stream>>>(ct, proj_w, proj_b, logits_w, logits_b,
                                                            hbuf, out + 1 + BATCH * HDIM);
  k_bnstats<<<HDIM, 256, 0, stream>>>(hbuf, mu, var);
  k_bnapply<<<(BATCH * HDIM) / 256, 256, 0, stream>>>(hbuf, mu, var, bn_gamma, bn_beta, out + 1);
  k_fin <<<1, 64, 0, stream>>>(nacc, out);
}
